// RGCN_31490700214799
// MI455X (gfx1250) — matrix operand order repaired and checked by reading, not compiled, not run on hardware
//
#include <hip/hip_runtime.h>
#include <stdint.h>

#define NN 8192
#define DD 256

typedef __attribute__((ext_vector_type(8)))  float          v8f;
typedef __attribute__((ext_vector_type(16))) __bf16         v16bf;
typedef __attribute__((ext_vector_type(8)))  unsigned int   v8u;
typedef __attribute__((ext_vector_type(4)))  unsigned int   u32x4;
typedef __attribute__((ext_vector_type(2)))  unsigned int   u32x2;
typedef __attribute__((ext_vector_type(4)))  float          f32x4;

// ---- fp32 -> bf16 (round to nearest even) helpers -------------------------
static __device__ __forceinline__ unsigned short f2bf(float f) {
  unsigned int u = __float_as_uint(f);
  u = (u + 0x7FFFu + ((u >> 16) & 1u)) >> 16;
  return (unsigned short)u;
}
static __device__ __forceinline__ unsigned int f2bf2(float a, float b) {
  return (unsigned int)f2bf(a) | ((unsigned int)f2bf(b) << 16);
}

// ---- WMMA fragment loaders ------------------------------------------------
// A fragment (16x32 bf16): lane m=L&15, khalf=(L>>4)*8.
// VGPRs 0-3 hold k=khalf..khalf+7 (16B), VGPRs 4-7 hold k=khalf+16..+23 (16B @ +32B)
static __device__ __forceinline__ v16bf lds_afrag(const unsigned short* p) {
  v8u w;
  w.lo = *(const u32x4*)p;
  w.hi = *(const u32x4*)(p + 16);
  return __builtin_bit_cast(v16bf, w);
}
// B fragment (32x16 bf16): lane n=L&15, khalf=(L>>4)*8; k=khalf..+7 and khalf+16..+23.
static __device__ __forceinline__ v16bf ld_bfrag(const unsigned short* p) {
  v8u w;
  w.lo = *(const u32x4*)p;
  w.hi = *(const u32x4*)(p + 16);
  return __builtin_bit_cast(v16bf, w);
}
static __device__ __forceinline__ v16bf ones_frag() {
  v8u w = {0x3F803F80u, 0x3F803F80u, 0x3F803F80u, 0x3F803F80u,
           0x3F803F80u, 0x3F803F80u, 0x3F803F80u, 0x3F803F80u};
  return __builtin_bit_cast(v16bf, w);
}
static __device__ __forceinline__ v8f wmma_bf16(v16bf a, v16bf b, v8f c) {
  return __builtin_amdgcn_wmma_f32_16x16x32_bf16(false, a, false, b,
                                                 (short)0, c, false, false);
}

// ===========================================================================
// K1: transpose + convert the 9 weight matrices (256x256 fp32 [d][o])
//     into WT bf16 [o][d]. Slot 0 = W_loop, 1..4 = W_in[r], 5..8 = W_out[r].
// ===========================================================================
__global__ __launch_bounds__(256)
void k_wt(const float* __restrict__ Wl, const float* __restrict__ Wi,
          const float* __restrict__ Wo, unsigned short* __restrict__ WT) {
  int idx = blockIdx.x * 256 + threadIdx.x;   // 9*65536 total
  int mat = idx >> 16;
  int rem = idx & 65535;
  int d = rem >> 8;
  int o = rem & 255;
  const float* src;
  if (mat == 0)      src = Wl;
  else if (mat <= 4) src = Wi + (size_t)(mat - 1) * 65536;
  else               src = Wo + (size_t)(mat - 5) * 65536;
  WT[(size_t)mat * 65536 + o * 256 + d] = f2bf(src[d * 256 + o]);
}

// ===========================================================================
// K2: H = X @ W for all 9 matrices via WMMA.
//   mat 0 -> d_out (fp32, full overwrite)   [X @ W_loop]
//   mat 1..4 -> HTin[r][o][j]  (bf16, transposed)
//   mat 5..8 -> HTout[r][o][j] (bf16, transposed)
// One workgroup = 16 rows of X; 8 waves each own 32 output columns.
// ===========================================================================
__global__ __launch_bounds__(256)
void k_feat(const float* __restrict__ X, const unsigned short* __restrict__ WT,
            unsigned short* __restrict__ HTin, unsigned short* __restrict__ HTout,
            float* __restrict__ out) {
  const int j0   = blockIdx.x * 16;
  const int tid  = threadIdx.x;
  const int lane = tid & 31;
  const int wave = tid >> 5;
  const int n0   = wave * 32;
  const int h    = lane >> 4;
  const int nn   = lane & 15;
  __shared__ __attribute__((aligned(16))) unsigned short sm[16 * 264]; // 16 rows x 256 k, pad to 264

  { // stage full 16x256 X strip as bf16
    const int row = tid >> 4, c4 = (tid & 15) * 4;
    for (int ro = 0; ro < 4; ++ro) {
      f32x4 x4 = *(const f32x4*)(X + (size_t)(j0 + row) * DD + ro * 64 + c4);
      u32x2 pk; pk.x = f2bf2(x4.x, x4.y); pk.y = f2bf2(x4.z, x4.w);
      *(u32x2*)&sm[row * 264 + ro * 64 + c4] = pk;
    }
  }
  __syncthreads();

  for (int mat = 0; mat < 9; ++mat) {
    v8f a0 = {}, a1 = {};
    const unsigned short* wb = WT + (size_t)mat * 65536;
    #pragma unroll
    for (int k32 = 0; k32 < 8; ++k32) {
      v16bf af = lds_afrag(sm + nn * 264 + k32 * 32 + h * 8);
      const int kb = k32 * 32 + h * 8;
      v16bf b0 = ld_bfrag(wb + (size_t)(n0 + nn) * 256 + kb);
      v16bf b1 = ld_bfrag(wb + (size_t)(n0 + 16 + nn) * 256 + kb);
      a0 = wmma_bf16(af, b0, a0);
      a1 = wmma_bf16(af, b1, a1);
    }
    const int mh = h * 8;
    if (mat == 0) {
      #pragma unroll
      for (int t = 0; t < 8; ++t) {
        out[(size_t)(j0 + mh + t) * DD + n0 + nn]      = a0[t];
        out[(size_t)(j0 + mh + t) * DD + n0 + 16 + nn] = a1[t];
      }
    } else {
      unsigned short* HT = (mat <= 4) ? HTin : HTout;
      const int r = (mat <= 4) ? (mat - 1) : (mat - 5);
      #pragma unroll
      for (int t = 0; t < 8; ++t) {
        HT[((size_t)r * DD + n0 + nn)      * NN + j0 + mh + t] = f2bf(a0[t]);
        HT[((size_t)r * DD + n0 + 16 + nn) * NN + j0 + mh + t] = f2bf(a1[t]);
      }
    }
  }
}

// ===========================================================================
// K3: incoming pass.  out[i0..i0+31] += sum_r (1/rowsum) * A[r,i,:] @ H_in[r]
// One workgroup owns a 32-row strip of A across all 8192 columns, loops r.
// Two 16-row m-tiles per k-chunk: each B fragment feeds 2 WMMAs (halves L2
// traffic vs 16-row strips). Double-buffered LDS tile (32x64 bf16, stride 72
// -> conflict-free ds_load_b128). Row sums via ones-WMMA (wave 0 only).
// ===========================================================================
__global__ __launch_bounds__(256)
void k_spmm_in(const float* __restrict__ A, const unsigned short* __restrict__ HT,
               float* __restrict__ out) {
  const int i0   = blockIdx.x * 32;
  const int tid  = threadIdx.x;
  const int lane = tid & 31;
  const int wave = tid >> 5;
  const int n0   = wave * 32;
  const int h    = lane >> 4;
  const int nn   = lane & 15;
  __shared__ __attribute__((aligned(16))) unsigned short sm[2][32 * 72];
  __shared__ float s_inv[32];

  const v16bf ones = ones_frag();
  v8f t00 = {}, t01 = {}, t10 = {}, t11 = {};
  const int lrow = tid >> 4;          // 0..15 (loader covers lrow and lrow+16)
  const int lc4  = (tid & 15) * 4;

  for (int r = 0; r < 4; ++r) {
    v8f a00 = {}, a01 = {}, a10 = {}, a11 = {}, as0 = {}, as1 = {};
    const float* Ar = A + (size_t)r * NN * NN + (size_t)i0 * NN;
    const unsigned short* HTr = HT + (size_t)r * DD * NN;
    { // preload round 0
      f32x4 v0 = *(const f32x4*)(Ar + (size_t)lrow * NN + lc4);
      f32x4 v1 = *(const f32x4*)(Ar + (size_t)(lrow + 16) * NN + lc4);
      u32x2 p0; p0.x = f2bf2(v0.x, v0.y); p0.y = f2bf2(v0.z, v0.w);
      u32x2 p1; p1.x = f2bf2(v1.x, v1.y); p1.y = f2bf2(v1.z, v1.w);
      *(u32x2*)&sm[0][lrow * 72 + lc4]        = p0;
      *(u32x2*)&sm[0][(lrow + 16) * 72 + lc4] = p1;
    }
    for (int ro = 0; ro < NN / 64; ++ro) {
      __syncthreads();
      if (ro + 1 < NN / 64) { // prefetch next tile into other buffer
        f32x4 v0 = *(const f32x4*)(Ar + (size_t)lrow * NN + (ro + 1) * 64 + lc4);
        f32x4 v1 = *(const f32x4*)(Ar + (size_t)(lrow + 16) * NN + (ro + 1) * 64 + lc4);
        u32x2 p0; p0.x = f2bf2(v0.x, v0.y); p0.y = f2bf2(v0.z, v0.w);
        u32x2 p1; p1.x = f2bf2(v1.x, v1.y); p1.y = f2bf2(v1.z, v1.w);
        unsigned short* b = sm[(ro + 1) & 1];
        *(u32x2*)&b[lrow * 72 + lc4]        = p0;
        *(u32x2*)&b[(lrow + 16) * 72 + lc4] = p1;
      }
      const unsigned short* buf = sm[ro & 1];
      #pragma unroll
      for (int c = 0; c < 2; ++c) {
        v16bf af0 = lds_afrag(buf + nn * 72 + c * 32 + h * 8);
        v16bf af1 = lds_afrag(buf + (nn + 16) * 72 + c * 32 + h * 8);
        const int kb = ro * 64 + c * 32 + h * 8;
        v16bf b0 = ld_bfrag(HTr + (size_t)(n0 + nn)      * NN + kb);
        v16bf b1 = ld_bfrag(HTr + (size_t)(n0 + 16 + nn) * NN + kb);
        a00 = wmma_bf16(af0, b0, a00);
        a01 = wmma_bf16(af0, b1, a01);
        a10 = wmma_bf16(af1, b0, a10);
        a11 = wmma_bf16(af1, b1, a11);
        if (wave == 0) {                     // degree sums (rows i0..i0+31)
          as0 = wmma_bf16(af0, ones, as0);
          as1 = wmma_bf16(af1, ones, as1);
        }
      }
    }
    __syncthreads();
    if (wave == 0 && nn == 0) {
      #pragma unroll
      for (int t = 0; t < 8; ++t) {
        float s0 = as0[t], s1 = as1[t];
        s_inv[h * 8 + t]      = (s0 == 0.0f) ? 1.0f : 1.0f / s0;
        s_inv[16 + h * 8 + t] = (s1 == 0.0f) ? 1.0f : 1.0f / s1;
      }
    }
    __syncthreads();
    #pragma unroll
    for (int t = 0; t < 8; ++t) {
      float i0v = s_inv[h * 8 + t];
      float i1v = s_inv[16 + h * 8 + t];
      t00[t] += a00[t] * i0v;
      t01[t] += a01[t] * i0v;
      t10[t] += a10[t] * i1v;
      t11[t] += a11[t] * i1v;
    }
    __syncthreads();
  }
  #pragma unroll
  for (int t = 0; t < 8; ++t) {
    size_t o0 = (size_t)(i0 + h * 8 + t) * DD;
    size_t o1 = (size_t)(i0 + 16 + h * 8 + t) * DD;
    out[o0 + n0 + nn]      += t00[t];
    out[o0 + n0 + 16 + nn] += t01[t];
    out[o1 + n0 + nn]      += t10[t];
    out[o1 + n0 + 16 + nn] += t11[t];
  }
}

// ===========================================================================
// K4: outgoing pass + fused ReLU.
// out[i] += sum_r (1/colsum[r,i]) * sum_j A[r,j,i] * H_out[r,j];  out = relu(out)
// Workgroup owns 32 columns of A[r]; tiles transposed through LDS so the WMMA
// path is identical to K3 (ones-WMMA now yields column sums).
// ===========================================================================
__global__ __launch_bounds__(256)
void k_spmm_out(const float* __restrict__ A, const unsigned short* __restrict__ HT,
                float* __restrict__ out) {
  const int i0   = blockIdx.x * 32;
  const int tid  = threadIdx.x;
  const int lane = tid & 31;
  const int wave = tid >> 5;
  const int n0   = wave * 32;
  const int h    = lane >> 4;
  const int nn   = lane & 15;
  __shared__ __attribute__((aligned(16))) unsigned short sm[2][32 * 72];
  __shared__ float s_inv[32];

  const v16bf ones = ones_frag();
  v8f t00 = {}, t01 = {}, t10 = {}, t11 = {};
  const int jl = tid >> 2;          // 0..63: source row of A
  const int i8 = (tid & 3) * 8;     // 8 columns of A -> 8 LDS rows

  for (int r = 0; r < 4; ++r) {
    v8f a00 = {}, a01 = {}, a10 = {}, a11 = {}, as0 = {}, as1 = {};
    const float* Ar = A + (size_t)r * NN * NN;
    const unsigned short* HTr = HT + (size_t)r * DD * NN;
    { // preload round 0 (transposed store)
      f32x4 v0 = *(const f32x4*)(Ar + (size_t)jl * NN + i0 + i8);
      f32x4 v1 = *(const f32x4*)(Ar + (size_t)jl * NN + i0 + i8 + 4);
      unsigned short* b = sm[0];
      b[(i8 + 0) * 72 + jl] = f2bf(v0.x);
      b[(i8 + 1) * 72 + jl] = f2bf(v0.y);
      b[(i8 + 2) * 72 + jl] = f2bf(v0.z);
      b[(i8 + 3) * 72 + jl] = f2bf(v0.w);
      b[(i8 + 4) * 72 + jl] = f2bf(v1.x);
      b[(i8 + 5) * 72 + jl] = f2bf(v1.y);
      b[(i8 + 6) * 72 + jl] = f2bf(v1.z);
      b[(i8 + 7) * 72 + jl] = f2bf(v1.w);
    }
    for (int ro = 0; ro < NN / 64; ++ro) {
      __syncthreads();
      if (ro + 1 < NN / 64) {
        f32x4 v0 = *(const f32x4*)(Ar + (size_t)((ro + 1) * 64 + jl) * NN + i0 + i8);
        f32x4 v1 = *(const f32x4*)(Ar + (size_t)((ro + 1) * 64 + jl) * NN + i0 + i8 + 4);
        unsigned short* b = sm[(ro + 1) & 1];
        b[(i8 + 0) * 72 + jl] = f2bf(v0.x);
        b[(i8 + 1) * 72 + jl] = f2bf(v0.y);
        b[(i8 + 2) * 72 + jl] = f2bf(v0.z);
        b[(i8 + 3) * 72 + jl] = f2bf(v0.w);
        b[(i8 + 4) * 72 + jl] = f2bf(v1.x);
        b[(i8 + 5) * 72 + jl] = f2bf(v1.y);
        b[(i8 + 6) * 72 + jl] = f2bf(v1.z);
        b[(i8 + 7) * 72 + jl] = f2bf(v1.w);
      }
      const unsigned short* buf = sm[ro & 1];
      #pragma unroll
      for (int c = 0; c < 2; ++c) {
        v16bf af0 = lds_afrag(buf + nn * 72 + c * 32 + h * 8);
        v16bf af1 = lds_afrag(buf + (nn + 16) * 72 + c * 32 + h * 8);
        const int kb = ro * 64 + c * 32 + h * 8;
        v16bf b0 = ld_bfrag(HTr + (size_t)(n0 + nn)      * NN + kb);
        v16bf b1 = ld_bfrag(HTr + (size_t)(n0 + 16 + nn) * NN + kb);
        a00 = wmma_bf16(af0, b0, a00);
        a01 = wmma_bf16(af0, b1, a01);
        a10 = wmma_bf16(af1, b0, a10);
        a11 = wmma_bf16(af1, b1, a11);
        if (wave == 0) {                     // column sums of A
          as0 = wmma_bf16(af0, ones, as0);
          as1 = wmma_bf16(af1, ones, as1);
        }
      }
    }
    __syncthreads();
    if (wave == 0 && nn == 0) {
      #pragma unroll
      for (int t = 0; t < 8; ++t) {
        float s0 = as0[t], s1 = as1[t];
        s_inv[h * 8 + t]      = (s0 == 0.0f) ? 1.0f : 1.0f / s0;
        s_inv[16 + h * 8 + t] = (s1 == 0.0f) ? 1.0f : 1.0f / s1;
      }
    }
    __syncthreads();
    #pragma unroll
    for (int t = 0; t < 8; ++t) {
      float i0v = s_inv[h * 8 + t];
      float i1v = s_inv[16 + h * 8 + t];
      t00[t] += a00[t] * i0v;
      t01[t] += a01[t] * i0v;
      t10[t] += a10[t] * i1v;
      t11[t] += a11[t] * i1v;
    }
    __syncthreads();
  }
  #pragma unroll
  for (int t = 0; t < 8; ++t) {
    size_t o0 = (size_t)(i0 + h * 8 + t) * DD;
    size_t o1 = (size_t)(i0 + 16 + h * 8 + t) * DD;
    out[o0 + n0 + nn]      = fmaxf(out[o0 + n0 + nn]      + t00[t], 0.0f);
    out[o0 + n0 + 16 + nn] = fmaxf(out[o0 + n0 + 16 + nn] + t01[t], 0.0f);
    out[o1 + n0 + nn]      = fmaxf(out[o1 + n0 + nn]      + t10[t], 0.0f);
    out[o1 + n0 + 16 + nn] = fmaxf(out[o1 + n0 + 16 + nn] + t11[t], 0.0f);
  }
}

// ===========================================================================
extern "C" void kernel_launch(void* const* d_in, const int* in_sizes, int n_in,
                              void* d_out, int out_size, void* d_ws, size_t ws_size,
                              hipStream_t stream) {
  (void)in_sizes; (void)n_in; (void)out_size; (void)ws_size;
  const float* X  = (const float*)d_in[0];   // [8192, 256]
  const float* A  = (const float*)d_in[1];   // [4, 8192, 8192]
  const float* Wl = (const float*)d_in[2];   // [256, 256]
  const float* Wi = (const float*)d_in[3];   // [4, 256, 256]
  const float* Wo = (const float*)d_in[4];   // [4, 256, 256]
  float* out = (float*)d_out;                // [8192, 256]

  // workspace layout (bytes): WT (9*64K bf16) | HTin (4*256*8192 bf16) | HTout
  unsigned short* WT    = (unsigned short*)d_ws;
  unsigned short* HTin  = (unsigned short*)((char*)d_ws + (size_t)9 * 65536 * 2);
  unsigned short* HTout = HTin + (size_t)4 * DD * NN;

  k_wt      <<<9 * 65536 / 256, 256, 0, stream>>>(Wl, Wi, Wo, WT);
  k_feat    <<<NN / 16,         256, 0, stream>>>(X, WT, HTin, HTout, out);
  k_spmm_in <<<NN / 32,         256, 0, stream>>>(A, HTin, out);
  k_spmm_out<<<NN / 32,         256, 0, stream>>>(A, HTout, out);
}
